// Quantizer_43026982371999
// MI455X (gfx1250) — matrix operand order repaired and checked by reading, not compiled, not run on hardware
//
#include <hip/hip_runtime.h>

typedef __attribute__((ext_vector_type(2)))  float  v2f;
typedef __attribute__((ext_vector_type(4)))  float  v4f;
typedef __attribute__((ext_vector_type(8)))  float  v8f;
typedef __attribute__((ext_vector_type(8)))  __bf16 v8bf;
typedef __attribute__((ext_vector_type(16))) __bf16 v16bf;

#define N_ROWS 32768
#define K_EMB  8192
#define D_DIM  256
#define LDSB_STRIDE_F32 260     // fp32 fallback path
#define LDSB_STRIDE_BF  264     // bf16 path: 256 + 8 halves pad (528 B rows)
#define LDS_ROW_BYTES   (LDSB_STRIDE_BF * 2)

// ===========================================================================
// bf16 split-precision path
// ===========================================================================

// ---- split z into hi/lo bf16 ----------------------------------------------
__global__ void vq_split_z_kernel(const float* __restrict__ z,
                                  __bf16* __restrict__ zhi,
                                  __bf16* __restrict__ zlo) {
    size_t i = ((size_t)blockIdx.x * 256 + threadIdx.x) * 4;
    v4f x = *(const v4f*)(z + i);
    v4f hf;
    __bf16 h0 = (__bf16)x.x, h1 = (__bf16)x.y, h2 = (__bf16)x.z, h3 = (__bf16)x.w;
    hf.x = (float)h0; hf.y = (float)h1; hf.z = (float)h2; hf.w = (float)h3;
    __bf16 l0 = (__bf16)(x.x - hf.x), l1 = (__bf16)(x.y - hf.y);
    __bf16 l2 = (__bf16)(x.z - hf.z), l3 = (__bf16)(x.w - hf.w);
    __attribute__((ext_vector_type(4))) __bf16 hv = {h0, h1, h2, h3};
    __attribute__((ext_vector_type(4))) __bf16 lv = {l0, l1, l2, l3};
    *(__attribute__((ext_vector_type(4))) __bf16*)(zhi + i) = hv;
    *(__attribute__((ext_vector_type(4))) __bf16*)(zlo + i) = lv;
}

// ---- split emb into hi/lo bf16 + row norms (one wave per row) -------------
__global__ void vq_split_emb_kernel(const float* __restrict__ emb,
                                    __bf16* __restrict__ ehi,
                                    __bf16* __restrict__ elo,
                                    float* __restrict__ enorm) {
    const int wave = threadIdx.x >> 5;
    const int lane = threadIdx.x & 31;
    const int row  = blockIdx.x * 8 + wave;       // 1024 blocks * 8 rows
    const size_t base = (size_t)row * D_DIM + lane * 8;
    v4f a = *(const v4f*)(emb + base);
    v4f b = *(const v4f*)(emb + base + 4);
    float s = a.x*a.x + a.y*a.y + a.z*a.z + a.w*a.w
            + b.x*b.x + b.y*b.y + b.z*b.z + b.w*b.w;
    float xs[8] = {a.x, a.y, a.z, a.w, b.x, b.y, b.z, b.w};
    v8bf hv, lv;
#pragma unroll
    for (int j = 0; j < 8; ++j) {
        __bf16 h = (__bf16)xs[j];
        hv[j] = h;
        lv[j] = (__bf16)(xs[j] - (float)h);
    }
    *(v8bf*)(ehi + base) = hv;
    *(v8bf*)(elo + base) = lv;
#pragma unroll
    for (int m = 16; m >= 1; m >>= 1) s += __shfl_xor(s, m, 32);
    if (lane == 0) enorm[row] = s;
}

// ---- async DMA of one 16x256 bf16 tile pair (hi+lo) into LDS --------------
// Each of the 256 threads issues 2x b128 per array: 512 lanes*16B = 8 KB/array.
// Per-lane offsets are loop-invariant; only the scalar global base changes.
__device__ __forceinline__ void issue_async_tile(const __bf16* gh, const __bf16* gl,
                                                 unsigned ldsH0, unsigned ldsH1,
                                                 unsigned ldsL0, unsigned ldsL1,
                                                 unsigned goff0, unsigned goff1) {
    asm volatile("global_load_async_to_lds_b128 %0, %1, %2"
                 :: "v"(ldsH0), "v"(goff0), "s"(gh) : "memory");
    asm volatile("global_load_async_to_lds_b128 %0, %1, %2"
                 :: "v"(ldsH1), "v"(goff1), "s"(gh) : "memory");
    asm volatile("global_load_async_to_lds_b128 %0, %1, %2"
                 :: "v"(ldsL0), "v"(goff0), "s"(gl) : "memory");
    asm volatile("global_load_async_to_lds_b128 %0, %1, %2"
                 :: "v"(ldsL1), "v"(goff1), "s"(gl) : "memory");
}

// ---- fused GEMM (3x bf16 WMMA 16x16x32) + argmin + gather -----------------
__launch_bounds__(256, 1)
__global__ void vq_main_bf16_kernel(const __bf16* __restrict__ zhi,
                                    const __bf16* __restrict__ zlo,
                                    const __bf16* __restrict__ ehi,
                                    const __bf16* __restrict__ elo,
                                    const float* __restrict__ enorm,
                                    const float* __restrict__ emb,
                                    float* __restrict__ out) {
    __shared__ __bf16 ldsH[2][16 * LDSB_STRIDE_BF];
    __shared__ __bf16 ldsL[2][16 * LDSB_STRIDE_BF];
    __shared__ int    lds_idx[128];

    const int tid  = threadIdx.x;
    const int wave = tid >> 5;
    const int lane = tid & 31;
    const int half = lane >> 4;
    const int l16  = lane & 15;
    const int r0   = blockIdx.x * 128;
    const int mrow = r0 + wave * 16 + l16;

    // Per-thread DMA addressing: chunk ids tid and tid+256 over 512 chunks
    // (16 rows x 32 sixteen-byte chunks). Row stride in LDS is 528 B (padded).
    const int c0id = tid, c1id = tid + 256;
    const int row0 = c0id >> 5, cb0 = (c0id & 31) << 4;
    const int row1 = c1id >> 5, cb1 = (c1id & 31) << 4;
    const unsigned goff0 = (unsigned)(row0 * 512 + cb0);
    const unsigned goff1 = (unsigned)(row1 * 512 + cb1);
    unsigned lH0[2], lH1[2], lL0[2], lL1[2];
#pragma unroll
    for (int b = 0; b < 2; ++b) {
        lH0[b] = (unsigned)(uintptr_t)((const char*)&ldsH[b][0] + row0 * LDS_ROW_BYTES + cb0);
        lH1[b] = (unsigned)(uintptr_t)((const char*)&ldsH[b][0] + row1 * LDS_ROW_BYTES + cb1);
        lL0[b] = (unsigned)(uintptr_t)((const char*)&ldsL[b][0] + row0 * LDS_ROW_BYTES + cb0);
        lL1[b] = (unsigned)(uintptr_t)((const char*)&ldsL[b][0] + row1 * LDS_ROW_BYTES + cb1);
    }

    // A panel resident: per k-step s (32 d-values), ISA 16-bit A layout:
    // lane<16: elems 0-7 = K 0..7, elems 8-15 = K 16..23 ; lane>=16: +8
    v16bf a_hi[8], a_lo[8];
    {
        const __bf16* zh = zhi + (size_t)mrow * D_DIM + 8 * half;
        const __bf16* zl = zlo + (size_t)mrow * D_DIM + 8 * half;
#pragma unroll
        for (int s = 0; s < 8; ++s) {
            v8bf h0 = *(const v8bf*)(zh + 32 * s);
            v8bf h1 = *(const v8bf*)(zh + 32 * s + 16);
            a_hi[s] = __builtin_shufflevector(h0, h1, 0,1,2,3,4,5,6,7,8,9,10,11,12,13,14,15);
            v8bf l0 = *(const v8bf*)(zl + 32 * s);
            v8bf l1 = *(const v8bf*)(zl + 32 * s + 16);
            a_lo[s] = __builtin_shufflevector(l0, l1, 0,1,2,3,4,5,6,7,8,9,10,11,12,13,14,15);
        }
    }

    float bestv[8];
    int   besti[8];
#pragma unroll
    for (int j = 0; j < 8; ++j) { bestv[j] = 3.4e38f; besti[j] = 0; }

    // Prologue: start DMA of tile 0 into buffer 0.
    issue_async_tile(ehi, elo, lH0[0], lH1[0], lL0[0], lL1[0], goff0, goff1);

    for (int c0 = 0; c0 < K_EMB; c0 += 16) {
        const int bi = (c0 >> 4) & 1;

        // Our DMA for this tile (issued last iteration) must have landed,
        // then barrier so every wave's DMA is visible before anyone reads.
        asm volatile("s_wait_asynccnt 0" ::: "memory");
        __syncthreads();

        // Kick off DMA for the next tile into the other buffer (that buffer
        // was last read in iteration c0-16, which everyone finished before
        // the barrier above).
        if (c0 + 16 < K_EMB) {
            const __bf16* gh = ehi + (size_t)(c0 + 16) * D_DIM;
            const __bf16* gl = elo + (size_t)(c0 + 16) * D_DIM;
            issue_async_tile(gh, gl, lH0[bi ^ 1], lH1[bi ^ 1],
                             lL0[bi ^ 1], lL1[bi ^ 1], goff0, goff1);
        }

        // ---- 8 k-steps x 3 WMMA (hi*hi, lo*hi, hi*lo), fp32 accumulate
        v8f acc_hh = {}, acc_hl = {}, acc_lh = {};
        const __bf16* bh = &ldsH[bi][l16 * LDSB_STRIDE_BF + 8 * half];
        const __bf16* bl = &ldsL[bi][l16 * LDSB_STRIDE_BF + 8 * half];
#pragma unroll
        for (int s = 0; s < 8; ++s) {
            v8bf b0 = *(const v8bf*)(bh + 32 * s);
            v8bf b1 = *(const v8bf*)(bh + 32 * s + 16);
            v16bf bhv = __builtin_shufflevector(b0, b1, 0,1,2,3,4,5,6,7,8,9,10,11,12,13,14,15);
            acc_hh = __builtin_amdgcn_wmma_f32_16x16x32_bf16(
                false, a_hi[s], false, bhv, (short)0, acc_hh, false, false);
            acc_lh = __builtin_amdgcn_wmma_f32_16x16x32_bf16(
                false, a_lo[s], false, bhv, (short)0, acc_lh, false, false);
            v8bf c0v = *(const v8bf*)(bl + 32 * s);
            v8bf c1v = *(const v8bf*)(bl + 32 * s + 16);
            v16bf blv = __builtin_shufflevector(c0v, c1v, 0,1,2,3,4,5,6,7,8,9,10,11,12,13,14,15);
            acc_hl = __builtin_amdgcn_wmma_f32_16x16x32_bf16(
                false, a_hi[s], false, blv, (short)0, acc_hl, false, false);
        }

        // ---- epilogue: dist = enorm[c] - 2*S ; running argmin
        const int   c  = c0 + l16;
        const float en = enorm[c];
#pragma unroll
        for (int j = 0; j < 8; ++j) {
            float dist = fmaf(-2.0f, acc_hh[j] + acc_hl[j] + acc_lh[j], en);
            if (dist < bestv[j]) { bestv[j] = dist; besti[j] = c; }
        }
    }

    // ---- reduce argmin across the 16 lanes sharing each row
#pragma unroll
    for (int j = 0; j < 8; ++j) {
#pragma unroll
        for (int m = 8; m >= 1; m >>= 1) {
            float ov = __shfl_xor(bestv[j], m, 32);
            int   oi = __shfl_xor(besti[j], m, 32);
            if (ov < bestv[j] || (ov == bestv[j] && oi < besti[j])) {
                bestv[j] = ov; besti[j] = oi;
            }
        }
    }
    if (l16 == 0) {
#pragma unroll
        for (int j = 0; j < 8; ++j)
            lds_idx[wave * 16 + half * 8 + j] = besti[j];
    }
    __syncthreads();

    // ---- fused gather from ORIGINAL fp32 emb
    {
        int rl = tid >> 1;
        int hh = tid & 1;
        int bi = lds_idx[rl];
        const v4f* src = (const v4f*)(emb + (size_t)bi * D_DIM + hh * 128);
        v4f*       dst = (v4f*)(out + (size_t)(r0 + rl) * D_DIM + hh * 128);
#pragma unroll
        for (int q = 0; q < 32; ++q) dst[q] = src[q];
    }
}

// ===========================================================================
// fp32 WMMA fallback path — used if workspace too small
// ===========================================================================
__global__ void vq_enorm_kernel(const float* __restrict__ emb,
                                float* __restrict__ enorm) {
    const int wave = threadIdx.x >> 5;
    const int lane = threadIdx.x & 31;
    const int row  = blockIdx.x * 8 + wave;
    const float* p = emb + (size_t)row * D_DIM + lane * 8;
    v4f a = *(const v4f*)p;
    v4f b = *(const v4f*)(p + 4);
    float s = a.x*a.x + a.y*a.y + a.z*a.z + a.w*a.w
            + b.x*b.x + b.y*b.y + b.z*b.z + b.w*b.w;
#pragma unroll
    for (int m = 16; m >= 1; m >>= 1) s += __shfl_xor(s, m, 32);
    if (lane == 0) enorm[row] = s;
}

__launch_bounds__(256, 1)
__global__ void vq_main_f32_kernel(const float* __restrict__ z,
                                   const float* __restrict__ emb,
                                   const float* __restrict__ enorm,
                                   float* __restrict__ out) {
    __shared__ float ldsB[16 * LDSB_STRIDE_F32];
    __shared__ int   lds_idx[128];
    const int tid  = threadIdx.x;
    const int wave = tid >> 5;
    const int lane = tid & 31;
    const int half = lane >> 4;
    const int l16  = lane & 15;
    const int r0   = blockIdx.x * 128;
    const int mrow = r0 + wave * 16 + l16;

    v2f a_reg[64];
    {
        const float* zp = z + (size_t)mrow * D_DIM + 2 * half;
#pragma unroll
        for (int k = 0; k < 64; ++k) a_reg[k] = *(const v2f*)(zp + 4 * k);
    }
    float bestv[8]; int besti[8];
#pragma unroll
    for (int j = 0; j < 8; ++j) { bestv[j] = 3.4e38f; besti[j] = 0; }

    for (int c0 = 0; c0 < K_EMB; c0 += 16) {
#pragma unroll
        for (int p = 0; p < 4; ++p) {
            int idx = p * 256 + tid;
            int brow = idx >> 6;
            int dp   = (idx & 63) << 2;
            *(v4f*)(&ldsB[brow * LDSB_STRIDE_F32 + dp]) =
                *(const v4f*)(emb + (size_t)(c0 + brow) * D_DIM + dp);
        }
        __syncthreads();
        v8f acc0 = {}; v8f acc1 = {};
        const float* bb = &ldsB[l16 * LDSB_STRIDE_F32 + 2 * half];
#pragma unroll
        for (int k = 0; k < 64; k += 2) {
            v2f b0 = *(const v2f*)(bb + 4 * k);
            acc0 = __builtin_amdgcn_wmma_f32_16x16x4_f32(
                false, a_reg[k], false, b0, (short)0, acc0, false, false);
            v2f b1 = *(const v2f*)(bb + 4 * (k + 1));
            acc1 = __builtin_amdgcn_wmma_f32_16x16x4_f32(
                false, a_reg[k + 1], false, b1, (short)0, acc1, false, false);
        }
        __syncthreads();
        const int   c  = c0 + l16;
        const float en = enorm[c];
#pragma unroll
        for (int j = 0; j < 8; ++j) {
            float dist = fmaf(-2.0f, acc0[j] + acc1[j], en);
            if (dist < bestv[j]) { bestv[j] = dist; besti[j] = c; }
        }
    }
#pragma unroll
    for (int j = 0; j < 8; ++j) {
#pragma unroll
        for (int m = 8; m >= 1; m >>= 1) {
            float ov = __shfl_xor(bestv[j], m, 32);
            int   oi = __shfl_xor(besti[j], m, 32);
            if (ov < bestv[j] || (ov == bestv[j] && oi < besti[j])) {
                bestv[j] = ov; besti[j] = oi;
            }
        }
    }
    if (l16 == 0) {
#pragma unroll
        for (int j = 0; j < 8; ++j) lds_idx[wave * 16 + half * 8 + j] = besti[j];
    }
    __syncthreads();
    {
        int rl = tid >> 1;
        int hh = tid & 1;
        int bi = lds_idx[rl];
        const v4f* src = (const v4f*)(emb + (size_t)bi * D_DIM + hh * 128);
        v4f*       dst = (v4f*)(out + (size_t)(r0 + rl) * D_DIM + hh * 128);
#pragma unroll
        for (int q = 0; q < 32; ++q) dst[q] = src[q];
    }
}

// ===========================================================================
extern "C" void kernel_launch(void* const* d_in, const int* in_sizes, int n_in,
                              void* d_out, int out_size, void* d_ws, size_t ws_size,
                              hipStream_t stream) {
    const float* z   = (const float*)d_in[0];   // [32768, 256] f32
    const float* emb = (const float*)d_in[1];   // [8192, 256]  f32
    float* out = (float*)d_out;

    const size_t enorm_bytes = (size_t)K_EMB * sizeof(float);            // 32 KB
    const size_t e_bytes     = (size_t)K_EMB * D_DIM * sizeof(__bf16);   // 4 MB
    const size_t z_bytes     = (size_t)N_ROWS * D_DIM * sizeof(__bf16);  // 16 MB
    const size_t need = enorm_bytes + 2 * e_bytes + 2 * z_bytes;         // ~40 MB

    char* ws = (char*)d_ws;
    float*  enorm = (float*)ws;
    __bf16* ehi   = (__bf16*)(ws + enorm_bytes);
    __bf16* elo   = (__bf16*)(ws + enorm_bytes + e_bytes);
    __bf16* zhi   = (__bf16*)(ws + enorm_bytes + 2 * e_bytes);
    __bf16* zlo   = (__bf16*)(ws + enorm_bytes + 2 * e_bytes + z_bytes);

    if (ws_size >= need) {
        vq_split_z_kernel<<<(N_ROWS * D_DIM / 4) / 256, 256, 0, stream>>>(z, zhi, zlo);
        vq_split_emb_kernel<<<K_EMB / 8, 256, 0, stream>>>(emb, ehi, elo, enorm);
        vq_main_bf16_kernel<<<N_ROWS / 128, 256, 0, stream>>>(zhi, zlo, ehi, elo,
                                                              enorm, emb, out);
    } else {
        vq_enorm_kernel<<<K_EMB / 8, 256, 0, stream>>>(emb, enorm);
        vq_main_f32_kernel<<<N_ROWS / 128, 256, 0, stream>>>(z, emb, enorm, out);
    }
}
